// NeuronMixtralDecoderLayer_12721693130957
// MI455X (gfx1250) — matrix operand order repaired and checked by reading, not compiled, not run on hardware
//
#include <hip/hip_runtime.h>
#include <cstddef>
#include <cstdint>

// ---------------------------------------------------------------------------
// MI455X (gfx1250) Mixtral decoder layer.
// Compute path: fp32 -> f16 -> v_wmma_f32_16x16x32_f16 (wave32).
// Weight stream (roofline-critical, 23.3 TB/s HBM): Tensor Data Mover
// (TENSOR_LOAD_TO_LDS) double-buffered into LDS, synced with s_wait_tensorcnt.
// ---------------------------------------------------------------------------

typedef __attribute__((ext_vector_type(16))) _Float16 v16h;
typedef __attribute__((ext_vector_type(8)))  _Float16 v8h;
typedef __attribute__((ext_vector_type(4)))  _Float16 v4h;
typedef __attribute__((ext_vector_type(8)))  float    v8f;
typedef __attribute__((ext_vector_type(4)))  unsigned v4u;
typedef __attribute__((ext_vector_type(8)))  int      v8i;
typedef __attribute__((ext_vector_type(4)))  int      v4i;

__device__ inline v8f wmma_f16(v16h a, v16h b, v8f c) {
  return __builtin_amdgcn_wmma_f32_16x16x32_f16(
      /*neg_a=*/false, a, /*neg_b=*/false, b,
      /*c_mod=*/(short)0, c, /*reuse_a=*/false, /*reuse_b=*/false);
}

__device__ inline v8f v8f_zero() {
  v8f z = {0.f, 0.f, 0.f, 0.f, 0.f, 0.f, 0.f, 0.f};
  return z;
}

// Load a 16-element f16 fragment from two aligned 8-half runs in LDS.
__device__ inline v16h ld_frag(const _Float16* p0, const _Float16* p1) {
  v8h lo = *(const v8h*)p0;
  v8h hi = *(const v8h*)p1;
  v16h r;
#pragma unroll
  for (int i = 0; i < 8; ++i) { r[i] = lo[i]; r[i + 8] = hi[i]; }
  return r;
}

// Pack 16 contiguous fp32 -> f16 fragment (B-matrix layout helper).
__device__ inline v16h pack16f(const float* p) {
  v16h r;
#pragma unroll
  for (int i = 0; i < 16; ++i) r[i] = (_Float16)p[i];
  return r;
}

// A-matrix fragment from a global fp32 row: elems 0..7 = row[kb..kb+7],
// elems 8..15 = row[kb+8..kb+15], scaled.
__device__ inline v16h packA_f32(const float* row, int kb, float sc) {
  v16h r;
#pragma unroll
  for (int i = 0; i < 8; ++i) {
    r[i]     = (_Float16)(row[kb + i] * sc);
    r[i + 8] = (_Float16)(row[kb + 8 + i] * sc);
  }
  return r;
}

// ---------------------------------------------------------------------------
// TDM: DMA one 32(k) x 64(n) fp32 weight tile (row stride N) into LDS, with
// 1-DWORD padding every 64 DWORDs -> LDS row stride 65 floats (bank-friendly).
// D# layout per CDNA5 ISA. Issued by one wave.
// ---------------------------------------------------------------------------
#define BM 128
#define BN 64
#define BK 32
#define ASTR 40       // halves per f16 LDS row (32 + 8 pad), 16B-aligned stride
#define BROW 65       // fp32 LDS row stride for TDM-staged B tile

__device__ inline void tdm_load_b_tile(const float* Bg, int N, int K,
                                       int gn0, int kb, float* ldsDst) {
  const unsigned lds_base = (unsigned)(uintptr_t)ldsDst;  // low 32b = LDS offset
  const unsigned long long ga =
      (unsigned long long)(uintptr_t)Bg +
      ((unsigned long long)(unsigned)kb * (unsigned)N + (unsigned)gn0) * 4ull;
  v4u g0;
  g0[0] = 1u;                                   // count=1, user descriptor
  g0[1] = lds_base;                             // lds_addr
  g0[2] = (unsigned)ga;                         // global_addr[31:0]
  g0[3] = (unsigned)(ga >> 32) | (2u << 30);    // global_addr[56:32] | type=2
  v8i g1;
  g1[0] = (int)((2u << 16) |                    // data_size = 4B
                (1u << 20) |                    // pad_enable
                (5u << 22));                    // pad_interval: 64 DWORDs
                                                // pad_amount = 0 -> 1 DWORD
  g1[1] = (int)(((unsigned)N & 0xFFFFu) << 16);              // tensor_dim0 lo16
  g1[2] = (int)(((unsigned)N >> 16) |
                (((unsigned)K & 0xFFFFu) << 16));            // dim0 hi | dim1 lo
  g1[3] = (int)(((unsigned)K >> 16) | (64u << 16));          // dim1 hi | tile_dim0
  g1[4] = 32;                                                // tile_dim1
  g1[5] = (int)N;                                            // dim0_stride lo32
  g1[6] = 0;                                                 // stride hi | dim1_stride lo
  g1[7] = 0;
  v4i gz; gz[0] = 0; gz[1] = 0; gz[2] = 0; gz[3] = 0;
#if defined(__clang_major__) && (__clang_major__ >= 23)
  v8i gz8; gz8[0]=0; gz8[1]=0; gz8[2]=0; gz8[3]=0; gz8[4]=0; gz8[5]=0; gz8[6]=0; gz8[7]=0;
  __builtin_amdgcn_tensor_load_to_lds(g0, g1, gz, gz, gz8, 0);
#else
  __builtin_amdgcn_tensor_load_to_lds(g0, g1, gz, gz, 0);
#endif
}

// ---------------------------------------------------------------------------
// Tiled WMMA GEMM: C[M,N] = A[M,K] * B[K,N]   (fp32 in/out, f16 compute)
//   MODE 0: C = A*B
//   MODE 1: C = A*B + R            (fused residual add)
//   MODE 2: C += rowscale[m]*A*B   (fused MoE combine-weight accumulate)
// Block: 256 threads = 8 waves (2n x 4m), tile 128x64, K-step 32.
// B (weights) staged by TDM, double-buffered; A staged as f16 by the waves.
// ---------------------------------------------------------------------------
template <int MODE>
__global__ __launch_bounds__(256) void gemm_wmma_kernel(
    const float* __restrict__ Ag, const float* __restrict__ Bg,
    float* __restrict__ Cg, const float* __restrict__ Rg,
    const float* __restrict__ rowscale, int rs_stride,
    int M, int N, int K) {
  __shared__ __align__(16) _Float16 As[BM * ASTR];
  __shared__ __align__(16) float    Bsf[2][BK * BROW];

  const int gm0 = blockIdx.x * BM;
  const int gn0 = blockIdx.y * BN;
  const int tid = threadIdx.x;
  const int lane = tid & 31;
  const int l16 = lane & 15;
  const int lh  = lane >> 4;
  const int wid = tid >> 5;
  const int m0 = (wid & 3) * 32;  // wave m-offset inside block tile
  const int n0 = (wid >> 2) * 32; // wave n-offset inside block tile

  v8f acc[2][2];
#pragma unroll
  for (int i = 0; i < 2; ++i)
#pragma unroll
    for (int j = 0; j < 2; ++j) acc[i][j] = v8f_zero();

  const int nk = K / BK;
  if (wid == 0) tdm_load_b_tile(Bg, N, K, gn0, 0, &Bsf[0][0]);

  for (int it_k = 0; it_k < nk; ++it_k) {
    const int kb = it_k * BK;
    const int cur = it_k & 1;

    // --- stage A tile (128x32) fp32->f16 while the TDM B-tile is in flight ---
#pragma unroll
    for (int it = 0; it < 4; ++it) {
      int lin = tid + it * 256;          // 0..1023
      int row = lin >> 3;
      int c4  = lin & 7;
      const float4 f =
          *(const float4*)(Ag + (size_t)(gm0 + row) * K + kb + c4 * 4);
      v4h h;
      h[0] = (_Float16)f.x; h[1] = (_Float16)f.y;
      h[2] = (_Float16)f.z; h[3] = (_Float16)f.w;
      *(v4h*)&As[row * ASTR + c4 * 4] = h;
    }
    // prefetch the A stream for the following block (global_prefetch_b8)
    if (kb + BK < K)
      __builtin_prefetch(Ag + (size_t)(gm0 + (tid >> 1)) * K + kb + BK, 0, 1);

    // --- issue next TDM tile, then wait for the current one (in-order) ---
    if (wid == 0) {
      if (it_k + 1 < nk) {
        tdm_load_b_tile(Bg, N, K, gn0, kb + BK, &Bsf[cur ^ 1][0]);
        __builtin_amdgcn_s_wait_tensorcnt(1);
      } else {
        __builtin_amdgcn_s_wait_tensorcnt(0);
      }
    }
    __syncthreads();

    // --- one 16x16x32 WMMA step per sub-tile ---
    v16h af[2], bf[2];
#pragma unroll
    for (int i = 0; i < 2; ++i) {
      int m = m0 + i * 16 + l16;
      int ko = lh * 8;
      af[i] = ld_frag(&As[m * ASTR + ko], &As[m * ASTR + ko + 16]);
    }
    const float* bp = &Bsf[cur][0];
#pragma unroll
    for (int j = 0; j < 2; ++j) {
      int n = n0 + j * 16 + l16;
      v16h bv;
#pragma unroll
      for (int jj = 0; jj < 16; ++jj)
        bv[jj] = (_Float16)bp[(lh * 8 + (jj & 7) + (jj >> 3) * 16) * BROW + n];
      bf[j] = bv;
    }
#pragma unroll
    for (int i = 0; i < 2; ++i)
#pragma unroll
      for (int j = 0; j < 2; ++j) acc[i][j] = wmma_f16(af[i], bf[j], acc[i][j]);
    __syncthreads();
  }

  // --- epilogue: C layout row = e + 8*(lane>>4), col = lane&15 ---
#pragma unroll
  for (int i = 0; i < 2; ++i)
#pragma unroll
    for (int j = 0; j < 2; ++j) {
      int gcol = gn0 + n0 + j * 16 + l16;
#pragma unroll
      for (int e = 0; e < 8; ++e) {
        int gm = gm0 + m0 + i * 16 + e + lh * 8;
        size_t idx = (size_t)gm * N + gcol;
        float val = acc[i][j][e];
        if (MODE == 0) {
          Cg[idx] = val;
        } else if (MODE == 1) {
          Cg[idx] = val + Rg[idx];
        } else {
          Cg[idx] += rowscale[(size_t)gm * rs_stride] * val;
        }
      }
    }
}

// ---------------------------------------------------------------------------
// RMSNorm: one block (256 thr) per row of D=2048.
// ---------------------------------------------------------------------------
__global__ __launch_bounds__(256) void rmsnorm_kernel(
    const float* __restrict__ x, const float* __restrict__ w,
    float* __restrict__ y, int D) {
  const int row = blockIdx.x;
  const float* xr = x + (size_t)row * D;
  float ss = 0.f;
  for (int i = threadIdx.x; i < D; i += 256) { float t = xr[i]; ss += t * t; }
#pragma unroll
  for (int off = 16; off > 0; off >>= 1) ss += __shfl_xor(ss, off, 32);
  __shared__ float red[8];
  if ((threadIdx.x & 31) == 0) red[threadIdx.x >> 5] = ss;
  __syncthreads();
  float tot = 0.f;
#pragma unroll
  for (int i = 0; i < 8; ++i) tot += red[i];
  const float inv = rsqrtf(tot / (float)D + 1e-5f);
  float* yr = y + (size_t)row * D;
  for (int i = threadIdx.x; i < D; i += 256) yr[i] = xr[i] * inv * w[i];
}

// ---------------------------------------------------------------------------
// RoPE (theta=1e6, HD=128): rotate pairs (i, i+64) in place.
// ---------------------------------------------------------------------------
__global__ void rope_kernel(float* __restrict__ x, const int* __restrict__ pos,
                            int T, int S_, int NH) {
  int idx = blockIdx.x * blockDim.x + threadIdx.x;
  int total = T * NH * 64;
  if (idx >= total) return;
  int i = idx & 63;
  int hh = (idx >> 6) % NH;
  int t = idx / (64 * NH);
  float p = (float)pos[t % S_];
  float inv_freq = __powf(1.0e6f, -(float)(2 * i) / 128.0f);
  float ang = p * inv_freq;
  float cs = __cosf(ang), sn = __sinf(ang);
  float* ptr = x + ((size_t)t * NH + hh) * 128;
  float a = ptr[i], b = ptr[i + 64];
  ptr[i]      = a * cs - b * sn;
  ptr[i + 64] = b * cs + a * sn;
}

// ---------------------------------------------------------------------------
// Flash attention: 1 wave per 16-query tile, causal+pad mask, GQA (H/KVH=4).
// QK^T and P*V both via v_wmma_f32_16x16x32_f16; P transposed through LDS.
// ---------------------------------------------------------------------------
__global__ __launch_bounds__(32) void flash_attn_kernel(
    const float* __restrict__ q, const float* __restrict__ k,
    const float* __restrict__ v, const float* __restrict__ amask,
    float* __restrict__ o, int S_) {
  constexpr int Hh = 16, KVHh = 4, HDh = 128;
  const int lane = threadIdx.x;
  const int l16 = lane & 15;
  const int lh  = lane >> 4;
  const int qt = blockIdx.x, h = blockIdx.y, b = blockIdx.z;
  const int kvh = h >> 2;
  const float sc = 0.08838834764831845f;  // 1/sqrt(128)

  // Q fragments (softmax scale folded in): 4 chunks of K=32 over HD=128
  const int qrow_s = qt * 16 + l16;
  const float* qrow = q + ((size_t)(b * S_ + qrow_s) * Hh + h) * HDh;
  v16h qa[4];
#pragma unroll
  for (int c = 0; c < 4; ++c) qa[c] = packA_f32(qrow, c * 32 + lh * 16, sc);

  v8f oacc[8];
#pragma unroll
  for (int c = 0; c < 8; ++c) oacc[c] = v8f_zero();
  float m_i[8], l_i[8];
#pragma unroll
  for (int e = 0; e < 8; ++e) { m_i[e] = -1e30f; l_i[e] = 0.f; }

  __shared__ __align__(16) _Float16 Ps[16 * ASTR];

  const int nkb = ((qt * 16 + 16) + 31) >> 5;  // key blocks of 32
  for (int ib = 0; ib < nkb; ++ib) {
    const int kb = ib * 32;
    // ---- scores S = Q K^T for 2 key tiles of 16 ----
    v8f s0 = v8f_zero(), s1 = v8f_zero();
    const int key0 = kb + l16, key1 = kb + 16 + l16;
#pragma unroll
    for (int c = 0; c < 4; ++c) {
      const int dbase = c * 32 + lh * 16;
      v16h b0 = pack16f(k + ((size_t)(b * S_ + key0) * KVHh + kvh) * HDh + dbase);
      v16h b1 = pack16f(k + ((size_t)(b * S_ + key1) * KVHh + kvh) * HDh + dbase);
      s0 = wmma_f16(qa[c], b0, s0);
      s1 = wmma_f16(qa[c], b1, s1);
    }
    // ---- mask + online softmax (row = e + 8*lh across the column group) ----
    const float pm0 = amask[(size_t)b * S_ + key0];
    const float pm1 = amask[(size_t)b * S_ + key1];
    float p0a[8], p1a[8];
#pragma unroll
    for (int e = 0; e < 8; ++e) {
      const int row = qt * 16 + e + lh * 8;
      float v0 = ((key0 <= row) && (pm0 > 0.5f)) ? s0[e] : -1e30f;
      float v1 = ((key1 <= row) && (pm1 > 0.5f)) ? s1[e] : -1e30f;
      float mx = fmaxf(v0, v1);
#pragma unroll
      for (int off = 1; off < 16; off <<= 1) mx = fmaxf(mx, __shfl_xor(mx, off, 32));
      const float mnew = fmaxf(m_i[e], mx);
      const float rescale = __expf(m_i[e] - mnew);
      const float e0 = __expf(v0 - mnew);
      const float e1 = __expf(v1 - mnew);
      float rs = e0 + e1;
#pragma unroll
      for (int off = 1; off < 16; off <<= 1) rs += __shfl_xor(rs, off, 32);
      l_i[e] = l_i[e] * rescale + rs;
      m_i[e] = mnew;
      p0a[e] = e0; p1a[e] = e1;
#pragma unroll
      for (int c = 0; c < 8; ++c) oacc[c][e] *= rescale;
    }
    // ---- transpose P through LDS into A-fragment layout ----
    __syncthreads();
#pragma unroll
    for (int e = 0; e < 8; ++e) {
      const int row = e + lh * 8;
      Ps[row * ASTR + l16]      = (_Float16)p0a[e];
      Ps[row * ASTR + 16 + l16] = (_Float16)p1a[e];
    }
    __syncthreads();
    const v16h pa = ld_frag(&Ps[l16 * ASTR + lh * 16], &Ps[l16 * ASTR + lh * 16 + 8]);
    // ---- O += P * V : 8 hd-chunks of 16 cols, K-dim = 32 keys ----
#pragma unroll
    for (int c = 0; c < 8; ++c) {
      v16h bv;
      const float* vb =
          v + ((size_t)(b * S_ + kb + lh * 16) * KVHh + kvh) * HDh + c * 16 + l16;
#pragma unroll
      for (int j = 0; j < 16; ++j) bv[j] = (_Float16)vb[(size_t)j * (KVHh * HDh)];
      oacc[c] = wmma_f16(pa, bv, oacc[c]);
    }
  }
  // ---- write O / l ----
#pragma unroll
  for (int c = 0; c < 8; ++c) {
#pragma unroll
    for (int e = 0; e < 8; ++e) {
      const int row = qt * 16 + e + lh * 8;
      const int col = c * 16 + l16;
      o[(size_t)(b * S_ + row) * (Hh * HDh) + h * HDh + col] = oacc[c][e] / l_i[e];
    }
  }
}

// ---------------------------------------------------------------------------
// Router: softmax over 8 experts, top-2, normalized combine weights. 1 wave/token.
// ---------------------------------------------------------------------------
__global__ __launch_bounds__(256) void router_kernel(
    const float* __restrict__ x, const float* __restrict__ rw,
    float* __restrict__ comb, int T, int D) {
  const int wid = threadIdx.x >> 5, lane = threadIdx.x & 31;
  const int t = blockIdx.x * 8 + wid;
  if (t >= T) return;
  float acc[8];
#pragma unroll
  for (int e = 0; e < 8; ++e) acc[e] = 0.f;
  const float* xr = x + (size_t)t * D;
  for (int d = lane; d < D; d += 32) {
    const float xv = xr[d];
    const float* rr = rw + (size_t)d * 8;
#pragma unroll
    for (int e = 0; e < 8; ++e) acc[e] += xv * rr[e];
  }
#pragma unroll
  for (int e = 0; e < 8; ++e)
#pragma unroll
    for (int off = 16; off > 0; off >>= 1) acc[e] += __shfl_xor(acc[e], off, 32);
  float mx = acc[0];
#pragma unroll
  for (int e = 1; e < 8; ++e) mx = fmaxf(mx, acc[e]);
  float pr[8], s = 0.f;
#pragma unroll
  for (int e = 0; e < 8; ++e) { pr[e] = __expf(acc[e] - mx); s += pr[e]; }
#pragma unroll
  for (int e = 0; e < 8; ++e) pr[e] /= s;
  int i1 = 0;
#pragma unroll
  for (int e = 1; e < 8; ++e) if (pr[e] > pr[i1]) i1 = e;
  int i2 = (i1 == 0) ? 1 : 0;
#pragma unroll
  for (int e = 0; e < 8; ++e) if (e != i1 && pr[e] > pr[i2]) i2 = e;
  const float nrm = pr[i1] + pr[i2];
  if (lane == 0) {
#pragma unroll
    for (int e = 0; e < 8; ++e)
      comb[(size_t)t * 8 + e] =
          (e == i1) ? pr[i1] / nrm : ((e == i2) ? pr[i2] / nrm : 0.f);
  }
}

// ---------------------------------------------------------------------------
// h = silu(g) * u  (in place into g)
// ---------------------------------------------------------------------------
__global__ __launch_bounds__(256) void silu_mul_kernel(
    float* __restrict__ g, const float* __restrict__ u, int n) {
  const int i = blockIdx.x * 256 + threadIdx.x;
  if (i < n) {
    const float x = g[i];
    g[i] = (x / (1.f + __expf(-x))) * u[i];
  }
}

// ---------------------------------------------------------------------------
extern "C" void kernel_launch(void* const* d_in, const int* in_sizes, int n_in,
                              void* d_out, int out_size, void* d_ws, size_t ws_size,
                              hipStream_t stream) {
  (void)in_sizes; (void)n_in; (void)out_size; (void)ws_size;
  constexpr int Bc = 4, Sc = 2048, Dc = 2048, Hc = 16, KVHc = 4, HDc = 128;
  constexpr int Ec = 8, Ic = 4096;
  constexpr int T = Bc * Sc;  // 8192 tokens

  const float* hidden = (const float*)d_in[0];
  const float* amask  = (const float*)d_in[1];
  const int*   posids = (const int*)d_in[2];
  const float* ln1w   = (const float*)d_in[3];
  const float* ln2w   = (const float*)d_in[4];
  const float* wq     = (const float*)d_in[5];
  const float* wk     = (const float*)d_in[6];
  const float* wv     = (const float*)d_in[7];
  const float* wo     = (const float*)d_in[8];
  const float* rw     = (const float*)d_in[9];
  const float* wgate  = (const float*)d_in[10];
  const float* wup    = (const float*)d_in[11];
  const float* wdown  = (const float*)d_in[12];
  float* out = (float*)d_out;

  // workspace carve-up (floats)
  float* ws = (float*)d_ws;
  size_t off = 0;
  float* xn     = ws + off; off += (size_t)T * Dc;          // norm output (reused)
  float* qb     = ws + off; off += (size_t)T * Hc * HDc;
  float* kb     = ws + off; off += (size_t)T * KVHc * HDc;
  float* vb     = ws + off; off += (size_t)T * KVHc * HDc;
  float* attn   = ws + off; off += (size_t)T * Hc * HDc;
  float* resid2 = ws + off; off += (size_t)T * Dc;
  float* comb   = ws + off; off += (size_t)T * Ec;
  float* gbuf   = ws + off; off += (size_t)T * Ic;
  float* ubuf   = ws + off; off += (size_t)T * Ic;

  const dim3 blk(256);

  // 1) RMSNorm 1
  rmsnorm_kernel<<<T, 256, 0, stream>>>(hidden, ln1w, xn, Dc);
  // 2) QKV projections (WMMA GEMMs, TDM weight streaming)
  gemm_wmma_kernel<0><<<dim3(T / BM, (Hc * HDc) / BN), blk, 0, stream>>>(
      xn, wq, qb, nullptr, nullptr, 0, T, Hc * HDc, Dc);
  gemm_wmma_kernel<0><<<dim3(T / BM, (KVHc * HDc) / BN), blk, 0, stream>>>(
      xn, wk, kb, nullptr, nullptr, 0, T, KVHc * HDc, Dc);
  gemm_wmma_kernel<0><<<dim3(T / BM, (KVHc * HDc) / BN), blk, 0, stream>>>(
      xn, wv, vb, nullptr, nullptr, 0, T, KVHc * HDc, Dc);
  // 3) RoPE on q and k
  {
    const int nq = T * Hc * 64;
    rope_kernel<<<(nq + 255) / 256, 256, 0, stream>>>(qb, posids, T, Sc, Hc);
    const int nk2 = T * KVHc * 64;
    rope_kernel<<<(nk2 + 255) / 256, 256, 0, stream>>>(kb, posids, T, Sc, KVHc);
  }
  // 4) Flash attention (wave32, WMMA)
  flash_attn_kernel<<<dim3(Sc / 16, Hc, Bc), 32, 0, stream>>>(
      qb, kb, vb, amask, attn, Sc);
  // 5) O-projection + residual (fused)
  gemm_wmma_kernel<1><<<dim3(T / BM, Dc / BN), blk, 0, stream>>>(
      attn, wo, resid2, hidden, nullptr, 0, T, Dc, Hc * HDc);
  // 6) RMSNorm 2
  rmsnorm_kernel<<<T, 256, 0, stream>>>(resid2, ln2w, xn, Dc);
  // 7) Router -> combine weights [T,E]
  router_kernel<<<(T + 7) / 8, 256, 0, stream>>>(xn, rw, comb, T, Dc);
  // 8) out = resid2 (second residual), then accumulate experts
  hipMemcpyAsync(out, resid2, (size_t)T * Dc * sizeof(float),
                 hipMemcpyDeviceToDevice, stream);
  for (int e = 0; e < Ec; ++e) {
    const float* wg = wgate + (size_t)e * Dc * Ic;
    const float* wu = wup   + (size_t)e * Dc * Ic;
    const float* wd = wdown + (size_t)e * Ic * Dc;
    gemm_wmma_kernel<0><<<dim3(T / BM, Ic / BN), blk, 0, stream>>>(
        xn, wg, gbuf, nullptr, nullptr, 0, T, Ic, Dc);
    gemm_wmma_kernel<0><<<dim3(T / BM, Ic / BN), blk, 0, stream>>>(
        xn, wu, ubuf, nullptr, nullptr, 0, T, Ic, Dc);
    silu_mul_kernel<<<(T * Ic) / 256, 256, 0, stream>>>(gbuf, ubuf, T * Ic);
    gemm_wmma_kernel<2><<<dim3(T / BM, Dc / BN), blk, 0, stream>>>(
        gbuf, wd, out, nullptr, comb + e, Ec, T, Dc, Ic);
  }
}
